// RGAT_60464549593285
// MI455X (gfx1250) — compile-verified, matrix operand order checked by reading, not run on hardware
//
#include <hip/hip_runtime.h>
#include <hip/hip_bf16.h>
#include <math.h>

// ---------------------------------------------------------------------------
// RGAT (2-layer GAT with edge weights) for MI455X / gfx1250.
// GEMMs use wave32 WMMA bf16 -> f32 (V_WMMA_F32_16X16X32_BF16), with native
// hardware f32->bf16 converts (v_cvt_*_bf16_f32) and b128 A-tile loads.
// Edge softmax/scatter phases use f32 atomics (L2-resident working set).
// ---------------------------------------------------------------------------

typedef __attribute__((ext_vector_type(16))) __bf16 v16bf;
typedef __attribute__((ext_vector_type(8)))  float  v8f;

#define NEG_SLOPE 0.2f

static __device__ __forceinline__ __bf16 f2bf(float f) {
  return (__bf16)f;   // native convert on gfx1250
}

// ---------------------------------------------------------------------------
// WMMA GEMM: H[nrows, wcols] = X[nrows, 64] @ W[64, wcols]   (wcols = 64 or 8)
// One wave handles a 16-row tile. K=64 split into two 16x16x32 WMMAs.
// A-fragment layout (ISA 7.12.2, 16-bit A 16x32): lanes 0-15 hold M=lane,
// elems [0..7]=K 0..7, [8..15]=K 16..23; lanes 16-31 hold M=lane-16,
// elems [0..7]=K 8..15, [8..15]=K 24..31. B mirrors with lanes = N columns.
// C/D layout: VGPR r -> (M = r or r+8 by lane half, N = lane&15).
// ---------------------------------------------------------------------------
__global__ void k_gemm_wmma(const float* __restrict__ X,
                            const float* __restrict__ W,
                            float* __restrict__ H,
                            int nrows, int wcols) {
  const int lane = threadIdx.x & 31;
  const int wave = threadIdx.x >> 5;
  const int tile = blockIdx.x * (blockDim.x >> 5) + wave;
  const int row0 = tile << 4;
  if (row0 >= nrows) return;               // wave-uniform: EXEC stays all-1s

  const int mi   = lane & 15;
  const int koff = (lane < 16) ? 0 : 8;
  int r = row0 + mi;
  if (r >= nrows) r = nrows - 1;           // clamp loads; stores are guarded
  const float* xrow = X + (size_t)r * 64u;
  __builtin_prefetch(xrow + 64, 0, 0);     // global_prefetch_b8 on next row

  // b128 loads: each K-chunk of 8 contiguous floats = two float4 loads
  const float4* xv = (const float4*)(xrow + koff);       // 16B-aligned (koff 0/8)
  float4 q0 = xv[0],  q1 = xv[1];                        // K = koff..koff+7
  float4 q2 = xv[4],  q3 = xv[5];                        // K = 16+koff..
  float4 q4 = xv[8],  q5 = xv[9];                        // K = 32+koff..
  float4 q6 = xv[12], q7 = xv[13];                       // K = 48+koff..

  v16bf a0, a1;
  a0[0]  = f2bf(q0.x); a0[1]  = f2bf(q0.y); a0[2]  = f2bf(q0.z); a0[3]  = f2bf(q0.w);
  a0[4]  = f2bf(q1.x); a0[5]  = f2bf(q1.y); a0[6]  = f2bf(q1.z); a0[7]  = f2bf(q1.w);
  a0[8]  = f2bf(q2.x); a0[9]  = f2bf(q2.y); a0[10] = f2bf(q2.z); a0[11] = f2bf(q2.w);
  a0[12] = f2bf(q3.x); a0[13] = f2bf(q3.y); a0[14] = f2bf(q3.z); a0[15] = f2bf(q3.w);
  a1[0]  = f2bf(q4.x); a1[1]  = f2bf(q4.y); a1[2]  = f2bf(q4.z); a1[3]  = f2bf(q4.w);
  a1[4]  = f2bf(q5.x); a1[5]  = f2bf(q5.y); a1[6]  = f2bf(q5.z); a1[7]  = f2bf(q5.w);
  a1[8]  = f2bf(q6.x); a1[9]  = f2bf(q6.y); a1[10] = f2bf(q6.z); a1[11] = f2bf(q6.w);
  a1[12] = f2bf(q7.x); a1[13] = f2bf(q7.y); a1[14] = f2bf(q7.z); a1[15] = f2bf(q7.w);

  const int ntiles = (wcols + 15) >> 4;
  for (int nt = 0; nt < ntiles; ++nt) {
    const int col = (nt << 4) + mi;
    v16bf b0, b1;
#pragma unroll
    for (int i = 0; i < 8; ++i) {
      float w0 = 0.f, w1 = 0.f, w2 = 0.f, w3 = 0.f;
      if (col < wcols) {                      // zero-pad W2 (64x8) to 16 cols
        w0 = W[(koff + i) * wcols + col];
        w1 = W[(16 + koff + i) * wcols + col];
        w2 = W[(32 + koff + i) * wcols + col];
        w3 = W[(48 + koff + i) * wcols + col];
      }
      b0[i] = f2bf(w0); b0[i + 8] = f2bf(w1);
      b1[i] = f2bf(w2); b1[i + 8] = f2bf(w3);
    }
    v8f c = {};
    c = __builtin_amdgcn_wmma_f32_16x16x32_bf16(false, a0, false, b0,
                                                (short)0, c, false, false);
    c = __builtin_amdgcn_wmma_f32_16x16x32_bf16(false, a1, false, b1,
                                                (short)0, c, false, false);
#pragma unroll
    for (int i = 0; i < 8; ++i) {
      int row = row0 + ((lane < 16) ? i : (i + 8));
      if (row < nrows && col < wcols)
        H[(size_t)row * wcols + col] = c[i];
    }
  }
}

// --------------------------- segment / edge kernels ------------------------

__global__ void k_init(float* __restrict__ agg, float* __restrict__ mx,
                       float* __restrict__ den, long long nAgg, int nNodes) {
  long long i = (long long)blockIdx.x * blockDim.x + threadIdx.x;
  if (i < nAgg) agg[i] = 0.f;
  if (i < nNodes) { mx[i] = -INFINITY; den[i] = 0.f; }
}

__global__ void k_scores(const float* __restrict__ H,
                         const float* __restrict__ as,
                         const float* __restrict__ ad,
                         float* __restrict__ es, float* __restrict__ ed,
                         int n, int C) {
  int i = blockIdx.x * blockDim.x + threadIdx.x;
  if (i >= n) return;
  const float* h = H + (size_t)i * C;
  float s = 0.f, d = 0.f;
  for (int c = 0; c < C; ++c) { s += h[c] * as[c]; d += h[c] * ad[c]; }
  es[i] = s;
  ed[i] = d;
}

static __device__ __forceinline__ void atomicMaxF(float* addr, float v) {
  // valid with -inf init: signed-int max for v>=0, unsigned-int min for v<0
  if (v >= 0.f) atomicMax((int*)addr, __float_as_int(v));
  else          atomicMin((unsigned int*)addr, __float_as_uint(v));
}

__global__ void k_edge_logit_max(const float* __restrict__ es,
                                 const float* __restrict__ ed,
                                 const int* __restrict__ src,
                                 const int* __restrict__ dst,
                                 float* __restrict__ elog,
                                 float* __restrict__ mx, int nE) {
  int e = blockIdx.x * blockDim.x + threadIdx.x;
  if (e >= nE) return;
  float z = es[src[e]] + ed[dst[e]];
  z = (z > 0.f) ? z : NEG_SLOPE * z;         // leaky_relu
  elog[e] = z;
  atomicMaxF(&mx[dst[e]], z);
}

__global__ void k_edge_exp(float* __restrict__ elog,
                           const float* __restrict__ mx,
                           const int* __restrict__ dst,
                           float* __restrict__ den, int nE) {
  int e = blockIdx.x * blockDim.x + threadIdx.x;
  if (e >= nE) return;
  int d = dst[e];
  float ex = __expf(elog[e] - mx[d]);
  elog[e] = ex;                               // reuse buffer: logit -> exp
  atomicAdd(&den[d], ex);
}

__global__ void k_edge_scatter(const float* __restrict__ H,
                               const float* __restrict__ ex,
                               const float* __restrict__ den,
                               const float* __restrict__ ev,
                               const int* __restrict__ src,
                               const int* __restrict__ dst,
                               float* __restrict__ agg,
                               int nE, int C) {
  long long idx = (long long)blockIdx.x * blockDim.x + threadIdx.x;
  long long total = (long long)nE * C;
  if (idx >= total) return;
  int e = (int)(idx / C);
  int c = (int)(idx - (long long)e * C);
  int d = dst[e];
  float coef = ex[e] / (den[d] + 1e-16f) * ev[e];
  atomicAdd(&agg[(size_t)d * C + c], coef * H[(size_t)src[e] * C + c]);
}

__global__ void k_bias_relu(float* __restrict__ agg,
                            const float* __restrict__ b,
                            long long total, int C) {
  long long i = (long long)blockIdx.x * blockDim.x + threadIdx.x;
  if (i >= total) return;
  int c = (int)(i % C);
  float v = agg[i] + b[c];
  agg[i] = (v > 0.f) ? v : 0.f;
}

__global__ void k_bias_logsoftmax8(const float* __restrict__ agg,
                                   const float* __restrict__ b,
                                   float* __restrict__ out, int n) {
  int i = blockIdx.x * blockDim.x + threadIdx.x;
  if (i >= n) return;
  float v[8];
  float mx = -INFINITY;
#pragma unroll
  for (int c = 0; c < 8; ++c) {
    v[c] = agg[(size_t)i * 8 + c] + b[c];
    mx = fmaxf(mx, v[c]);
  }
  float s = 0.f;
#pragma unroll
  for (int c = 0; c < 8; ++c) s += __expf(v[c] - mx);
  float l = __logf(s) + mx;
#pragma unroll
  for (int c = 0; c < 8; ++c) out[(size_t)i * 8 + c] = v[c] - l;
}

// ------------------------------ launcher -----------------------------------

static inline int cdiv(long long a, long long b) { return (int)((a + b - 1) / b); }

extern "C" void kernel_launch(void* const* d_in, const int* in_sizes, int n_in,
                              void* d_out, int out_size, void* d_ws, size_t ws_size,
                              hipStream_t stream) {
  const float* x    = (const float*)d_in[0];
  const int*   ei   = (const int*)  d_in[1];
  const float* ev   = (const float*)d_in[2];
  const float* W1   = (const float*)d_in[3];
  const float* a1s  = (const float*)d_in[4];
  const float* a1d  = (const float*)d_in[5];
  const float* b1   = (const float*)d_in[6];
  const float* W2   = (const float*)d_in[7];
  const float* a2s  = (const float*)d_in[8];
  const float* a2d  = (const float*)d_in[9];
  const float* b2   = (const float*)d_in[10];

  const int N = in_sizes[0] / 64;       // x is [N, 64]
  const int E = in_sizes[1] / 2;        // edge_index is [2, E]
  const int* src = ei;
  const int* dst = ei + E;

  // workspace carve-out (floats)
  float* ws   = (float*)d_ws;
  float* h1   = ws;  ws += (size_t)N * 64;   // x @ W1
  float* agg1 = ws;  ws += (size_t)N * 64;   // layer-1 aggregated (then relu)
  float* h2   = ws;  ws += (size_t)N * 8;    // hid @ W2
  float* agg2 = ws;  ws += (size_t)N * 8;    // layer-2 aggregated
  float* es   = ws;  ws += N;
  float* ed   = ws;  ws += N;
  float* mx   = ws;  ws += N;
  float* den  = ws;  ws += N;
  float* elog = ws;  ws += E;                // per-edge logit, then exp

  const int B = 256;                         // 8 wave32s per block
  const int tilesN  = cdiv(N, 16);
  const int gemmGrd = cdiv(tilesN, B / 32);

  // ---------------- layer 1 (C = 64) ----------------
  k_init<<<cdiv((long long)N * 64, B), B, 0, stream>>>(agg1, mx, den, (long long)N * 64, N);
  k_gemm_wmma<<<gemmGrd, B, 0, stream>>>(x, W1, h1, N, 64);
  k_scores<<<cdiv(N, B), B, 0, stream>>>(h1, a1s, a1d, es, ed, N, 64);
  k_edge_logit_max<<<cdiv(E, B), B, 0, stream>>>(es, ed, src, dst, elog, mx, E);
  k_edge_exp<<<cdiv(E, B), B, 0, stream>>>(elog, mx, dst, den, E);
  k_edge_scatter<<<cdiv((long long)E * 64, B), B, 0, stream>>>(h1, elog, den, ev,
                                                               src, dst, agg1, E, 64);
  k_bias_relu<<<cdiv((long long)N * 64, B), B, 0, stream>>>(agg1, b1, (long long)N * 64, 64);

  // ---------------- layer 2 (C = 8) ----------------
  k_init<<<cdiv((long long)N * 8, B), B, 0, stream>>>(agg2, mx, den, (long long)N * 8, N);
  k_gemm_wmma<<<gemmGrd, B, 0, stream>>>(agg1, W2, h2, N, 8);
  k_scores<<<cdiv(N, B), B, 0, stream>>>(h2, a2s, a2d, es, ed, N, 8);
  k_edge_logit_max<<<cdiv(E, B), B, 0, stream>>>(es, ed, src, dst, elog, mx, E);
  k_edge_exp<<<cdiv(E, B), B, 0, stream>>>(elog, mx, dst, den, E);
  k_edge_scatter<<<cdiv((long long)E * 8, B), B, 0, stream>>>(h2, elog, den, ev,
                                                              src, dst, agg2, E, 8);
  k_bias_logsoftmax8<<<cdiv(N, B), B, 0, stream>>>(agg2, b2, (float*)d_out, N);
}
